// MultiHeadAttention_7679401525783
// MI455X (gfx1250) — compile-verified, matrix operand order checked by reading, not run on hardware
//
#include <hip/hip_runtime.h>
#include <math.h>

// ---------------------------------------------------------------------------
// MHA for MI455X (gfx1250, wave32, WMMA). bf16 data path, f32 accumulation.
//
// WMMA fragment layout (CDNA5 ISA 7.12.2, 16-bit A 16x32):
//   lane = l, half = l>>4, m/n = l&15
//   element j of v16bf maps to k(j) = 16*(j>>3) + 8*half + (j&7)
//   => per lane two contiguous 8-element (16B) runs at cols {8*half, 16+8*half}
// B operand stored transposed [N,K] row-major so it uses the same loader.
// C/D f32 frag: element r -> row (r + 8*half), col (lane&15).
// ---------------------------------------------------------------------------

typedef __attribute__((ext_vector_type(16))) __bf16 v16bf;
typedef __attribute__((ext_vector_type(8)))  __bf16 v8bf;
typedef __attribute__((ext_vector_type(8)))  float  v8f;

#define BB   2
#define TQL  2048
#define TKL  2048
#define HH   16
#define DHL  64
#define KD   1024   // X_TO_DIM == X_FROM_DIM == HIDDEN
#define ND   1024

static __device__ __forceinline__ v16bf load_frag(const __bf16* base, int ld,
                                                  int row, int col0, int half) {
  const __bf16* p = base + (size_t)row * ld + col0 + half * 8;
  v8bf lo = *(const v8bf*)(p);
  v8bf hi = *(const v8bf*)(p + 16);
  return __builtin_shufflevector(lo, hi, 0,1,2,3,4,5,6,7,8,9,10,11,12,13,14,15);
}

static __device__ __forceinline__ v8f wmma_bf16(v16bf a, v16bf b, v8f c) {
  return __builtin_amdgcn_wmma_f32_16x16x32_bf16(false, a, false, b,
                                                 (short)0, c, false, false);
}

// ---------------------------------------------------------------------------
// fp32 -> bf16 straight copy
__global__ void cvt_bf16_kernel(const float* __restrict__ x,
                                __bf16* __restrict__ y, int n) {
  int i = blockIdx.x * blockDim.x + threadIdx.x;
  if (i < n) y[i] = (__bf16)x[i];
}

// Reorganize per-head weight [H, K, DH] (f32) -> transposed bf16 [N=H*DH, K]
__global__ void reorg_w_kernel(const float* __restrict__ W,
                               __bf16* __restrict__ WT) {
  int i = blockIdx.x * blockDim.x + threadIdx.x;   // i over ND*KD
  int n = i >> 10;          // output column (h*64+d)
  int f = i & (KD - 1);     // input feature
  int h = n >> 6, d = n & 63;
  WT[(size_t)n * KD + f] = (__bf16)W[((size_t)h * KD + f) * DHL + d];
}

// ---------------------------------------------------------------------------
// WMMA GEMM: C[M,1024] = A[M,1024](bf16) x BT[1024,1024](bf16,[N,K]) + bias
// mode 0: Q  -> bf16 [b,h,t,d], scaled by 1/sqrt(DH)
// mode 1: K  -> bf16 [b,h,t,d]
// mode 2: V  -> bf16 transposed [b,h,d,t]
// mode 3: out-> f32  [row, n]
// Block = 128 threads (4 waves); workgroup tile 128x64; wave tile 32x64.
// Inner loop is double-buffered so loads overlap WMMAs (partial loadcnt waits).
__global__ __launch_bounds__(128)
void gemm_wmma_kernel(const __bf16* __restrict__ A, const __bf16* __restrict__ BT,
                      const float* __restrict__ bias,
                      __bf16* __restrict__ outb, float* __restrict__ outf,
                      int mode) {
  const int lane = threadIdx.x & 31;
  const int wave = threadIdx.x >> 5;
  const int half = lane >> 4;
  const int ln   = lane & 15;

  const int row0 = blockIdx.x * 128 + wave * 32;   // 2 row tiles of 16
  const int col0 = blockIdx.y * 64;                // 4 col tiles of 16

  v8f acc[2][4] = {};

  v16bf a0[2], b0[4], a1[2], b1[4];

  // prologue: k = 0
#pragma unroll
  for (int rt = 0; rt < 2; ++rt) a0[rt] = load_frag(A, KD, row0 + rt * 16 + ln, 0, half);
#pragma unroll
  for (int nt = 0; nt < 4; ++nt) b0[nt] = load_frag(BT, KD, col0 + nt * 16 + ln, 0, half);

  for (int k = 0; k < KD; k += 64) {
    // issue loads for k+32 into the other buffer, then do math on buffer 0
#pragma unroll
    for (int rt = 0; rt < 2; ++rt) a1[rt] = load_frag(A, KD, row0 + rt * 16 + ln, k + 32, half);
#pragma unroll
    for (int nt = 0; nt < 4; ++nt) b1[nt] = load_frag(BT, KD, col0 + nt * 16 + ln, k + 32, half);
#pragma unroll
    for (int rt = 0; rt < 2; ++rt)
#pragma unroll
      for (int nt = 0; nt < 4; ++nt) acc[rt][nt] = wmma_bf16(a0[rt], b0[nt], acc[rt][nt]);

    if (k + 64 < KD) {   // refill buffer 0 with k+64 while buffer 1 computes
#pragma unroll
      for (int rt = 0; rt < 2; ++rt) a0[rt] = load_frag(A, KD, row0 + rt * 16 + ln, k + 64, half);
#pragma unroll
      for (int nt = 0; nt < 4; ++nt) b0[nt] = load_frag(BT, KD, col0 + nt * 16 + ln, k + 64, half);
    }
#pragma unroll
    for (int rt = 0; rt < 2; ++rt)
#pragma unroll
      for (int nt = 0; nt < 4; ++nt) acc[rt][nt] = wmma_bf16(a1[rt], b1[nt], acc[rt][nt]);
  }

#pragma unroll
  for (int rt = 0; rt < 2; ++rt) {
#pragma unroll
    for (int nt = 0; nt < 4; ++nt) {
      const int col = col0 + nt * 16 + ln;
      const float bv = bias[col];
#pragma unroll
      for (int r = 0; r < 8; ++r) {
        const int grow = row0 + rt * 16 + r + 8 * half;  // global row in [0,4096)
        float v = acc[rt][nt][r] + bv;
        if (mode == 3) {
          outf[(size_t)grow * ND + col] = v;
        } else {
          const int b = grow >> 11, t = grow & (TQL - 1);
          const int h = col >> 6,  d = col & 63;
          if (mode == 0) v *= 0.125f;                    // 1/sqrt(64) folded into Q
          if (mode == 2) {                               // V stored transposed [d, s]
            outb[(((size_t)(b * HH + h)) * DHL + d) * TKL + t] = (__bf16)v;
          } else {
            outb[(((size_t)(b * HH + h)) * TQL + t) * DHL + d] = (__bf16)v;
          }
        }
      }
    }
  }
}

// ---------------------------------------------------------------------------
// Flash attention: one wave per 32-row Q tile, online softmax over TK.
// Qb: [bh, t, d] bf16 (pre-scaled), Kb: [bh, s, d] bf16, Vtb: [bh, d, s] bf16
// CC: [b, t, h*64+d] bf16 (concat layout, feeds output projection).
// Each K/V fragment is reused by both row tiles: 16 WMMAs per 8 frag loads.
__global__ __launch_bounds__(64)
void flash_attn_kernel(const __bf16* __restrict__ Qb, const __bf16* __restrict__ Kb,
                       const __bf16* __restrict__ Vtb, __bf16* __restrict__ CC) {
  const int lane = threadIdx.x & 31;
  const int wave = threadIdx.x >> 5;
  const int half = lane >> 4;
  const int ln   = lane & 15;

  const int bh = blockIdx.x;                 // b*H + h
  const int qt = blockIdx.y * 2 + wave;      // q tile index, 32 rows each
  const int row0 = qt * 32;

  const __bf16* Qh = Qb  + (size_t)bh * TQL * DHL;
  const __bf16* Kh = Kb  + (size_t)bh * TKL * DHL;
  const __bf16* Vh = Vtb + (size_t)bh * DHL * TKL;

  __shared__ __align__(16) __bf16 lds_p[2][32 * 32];  // per-wave P scratch
  __bf16* P = lds_p[wave];

  // Q fragments held for the whole K sweep (2 row tiles x 2 k-steps of 32)
  v16bf aq[2][2];
#pragma unroll
  for (int rt = 0; rt < 2; ++rt)
#pragma unroll
    for (int kk = 0; kk < 2; ++kk)
      aq[rt][kk] = load_frag(Qh, DHL, row0 + rt * 16 + ln, kk * 32, half);

  v8f o[2][4] = {};
  float mrow[2][8], lrow[2][8];
#pragma unroll
  for (int rt = 0; rt < 2; ++rt)
#pragma unroll
    for (int r = 0; r < 8; ++r) { mrow[rt][r] = -INFINITY; lrow[rt][r] = 0.0f; }

  for (int s0 = 0; s0 < TKL; s0 += 32) {
    // get next block's K/V lines moving while we do softmax VALU work
    if (s0 + 32 < TKL) {
      __builtin_prefetch(Kh + (size_t)(s0 + 32 + ln) * DHL, 0, 0);
      __builtin_prefetch(Vh + (size_t)(ln << 2) * TKL + s0 + 32, 0, 0);
    }

    // S tiles [2 x 16 x 32] = Q[32x64] x K^T ; K frags shared across row tiles
    v8f s[2][2] = {};
#pragma unroll
    for (int t = 0; t < 2; ++t) {
#pragma unroll
      for (int kk = 0; kk < 2; ++kk) {
        v16bf bk = load_frag(Kh, DHL, s0 + t * 16 + ln, kk * 32, half);
#pragma unroll
        for (int rt = 0; rt < 2; ++rt) s[rt][t] = wmma_bf16(aq[rt][kk], bk, s[rt][t]);
      }
    }

    // Online softmax; rows live across 16-lane groups (xor 1,2,4,8 reduce)
#pragma unroll
    for (int rt = 0; rt < 2; ++rt) {
#pragma unroll
      for (int r = 0; r < 8; ++r) {
        float mx = fmaxf(s[rt][0][r], s[rt][1][r]);
        mx = fmaxf(mx, __shfl_xor(mx, 1, 32));
        mx = fmaxf(mx, __shfl_xor(mx, 2, 32));
        mx = fmaxf(mx, __shfl_xor(mx, 4, 32));
        mx = fmaxf(mx, __shfl_xor(mx, 8, 32));
        const float mnew = fmaxf(mrow[rt][r], mx);
        const float corr = __expf(mrow[rt][r] - mnew);
        const float p0 = __expf(s[rt][0][r] - mnew);
        const float p1 = __expf(s[rt][1][r] - mnew);
        float rs = p0 + p1;
        rs += __shfl_xor(rs, 1, 32);
        rs += __shfl_xor(rs, 2, 32);
        rs += __shfl_xor(rs, 4, 32);
        rs += __shfl_xor(rs, 8, 32);
        lrow[rt][r] = lrow[rt][r] * corr + rs;
        mrow[rt][r] = mnew;
#pragma unroll
        for (int d = 0; d < 4; ++d) o[rt][d][r] *= corr;
        // C-layout -> LDS (row-major 32x32); LDS is in-order within a wave
        P[(rt * 16 + r + 8 * half) * 32 + ln]      = (__bf16)p0;
        P[(rt * 16 + r + 8 * half) * 32 + 16 + ln] = (__bf16)p1;
      }
    }

    // Re-read P as A fragments, accumulate O += P x V ; V frags shared
    v16bf ap[2];
#pragma unroll
    for (int rt = 0; rt < 2; ++rt) ap[rt] = load_frag(P, 32, rt * 16 + ln, 0, half);
#pragma unroll
    for (int d = 0; d < 4; ++d) {
      v16bf bv = load_frag(Vh, TKL, d * 16 + ln, s0, half);
#pragma unroll
      for (int rt = 0; rt < 2; ++rt) o[rt][d] = wmma_bf16(ap[rt], bv, o[rt][d]);
    }
  }

  // Normalize and store in concat layout [b, t, h*64 + d]
  const int b = bh >> 4, h = bh & 15;
#pragma unroll
  for (int rt = 0; rt < 2; ++rt) {
#pragma unroll
    for (int r = 0; r < 8; ++r) {
      const int trow = row0 + rt * 16 + r + 8 * half;
      const float inv = 1.0f / lrow[rt][r];
#pragma unroll
      for (int d = 0; d < 4; ++d) {
        const int col = h * 64 + d * 16 + ln;
        CC[((size_t)b * TQL + trow) * ND + col] = (__bf16)(o[rt][d][r] * inv);
      }
    }
  }
}

// ---------------------------------------------------------------------------
extern "C" void kernel_launch(void* const* d_in, const int* in_sizes, int n_in,
                              void* d_out, int out_size, void* d_ws, size_t ws_size,
                              hipStream_t stream) {
  (void)in_sizes; (void)n_in; (void)out_size; (void)ws_size;

  const float* x_to   = (const float*)d_in[0];
  const float* x_from = (const float*)d_in[1];
  const float* Wq     = (const float*)d_in[2];
  const float* bq     = (const float*)d_in[3];
  const float* Wk     = (const float*)d_in[4];
  const float* bk     = (const float*)d_in[5];
  const float* Wv     = (const float*)d_in[6];
  const float* bv     = (const float*)d_in[7];
  const float* Wo     = (const float*)d_in[8];
  const float* bo     = (const float*)d_in[9];
  float* out = (float*)d_out;

  // Workspace carve-up (bf16 elements). Total ~29.4M elts ~= 58.7 MB.
  __bf16* ws = (__bf16*)d_ws;
  size_t off = 0;
  __bf16* xt_bf = ws + off; off += (size_t)BB * TQL * KD;   // 4096x1024
  __bf16* xf_bf = ws + off; off += (size_t)BB * TKL * KD;
  __bf16* wq_t  = ws + off; off += (size_t)ND * KD;         // [N,K]
  __bf16* wk_t  = ws + off; off += (size_t)ND * KD;
  __bf16* wv_t  = ws + off; off += (size_t)ND * KD;
  __bf16* wo_b  = ws + off; off += (size_t)ND * KD;         // Wo already [N,K]
  __bf16* Qb    = ws + off; off += (size_t)BB * HH * TQL * DHL;
  __bf16* Kb    = ws + off; off += (size_t)BB * HH * TKL * DHL;
  __bf16* Vtb   = ws + off; off += (size_t)BB * HH * DHL * TKL;
  __bf16* CCb   = ws + off; off += (size_t)BB * TQL * ND;

  const int nAct = BB * TQL * KD;     // 4,194,304
  const int nW   = ND * KD;           // 1,048,576

  // Stage 1: precision conversion / weight reorg
  cvt_bf16_kernel<<<(nAct + 255) / 256, 256, 0, stream>>>(x_to,   xt_bf, nAct);
  cvt_bf16_kernel<<<(nAct + 255) / 256, 256, 0, stream>>>(x_from, xf_bf, nAct);
  cvt_bf16_kernel<<<(nW   + 255) / 256, 256, 0, stream>>>(Wo,     wo_b,  nW);
  reorg_w_kernel <<<(nW   + 255) / 256, 256, 0, stream>>>(Wq, wq_t);
  reorg_w_kernel <<<(nW   + 255) / 256, 256, 0, stream>>>(Wk, wk_t);
  reorg_w_kernel <<<(nW   + 255) / 256, 256, 0, stream>>>(Wv, wv_t);

  // Stage 2: QKV projections (WMMA GEMMs), M=4096, N=K=1024
  dim3 gg(32, 16), gb(128);
  gemm_wmma_kernel<<<gg, gb, 0, stream>>>(xt_bf, wq_t, bq, Qb,  nullptr, 0);
  gemm_wmma_kernel<<<gg, gb, 0, stream>>>(xf_bf, wk_t, bk, Kb,  nullptr, 1);
  gemm_wmma_kernel<<<gg, gb, 0, stream>>>(xf_bf, wv_t, bv, Vtb, nullptr, 2);

  // Stage 3: fused flash attention (never materializes [TQ,TK] scores)
  flash_attn_kernel<<<dim3(BB * HH, TQL / 64), 64, 0, stream>>>(Qb, Kb, Vtb, CCb);

  // Stage 4: output projection -> f32 result
  gemm_wmma_kernel<<<gg, gb, 0, stream>>>(CCb, wo_b, bo, nullptr, out, 3);
}
